// RecursiveCubeAttention_42245298323785
// MI455X (gfx1250) — matrix operand order repaired and checked by reading, not compiled, not run on hardware
//
#include <hip/hip_runtime.h>

// ---------- types ----------
typedef __attribute__((ext_vector_type(16))) __bf16          v16bf;
typedef __attribute__((ext_vector_type(8)))  float           v8f;
typedef __attribute__((ext_vector_type(8)))  float           f8v;
typedef __attribute__((ext_vector_type(8)))  unsigned short  us8;
typedef __attribute__((ext_vector_type(4)))  unsigned short  us4;
typedef __attribute__((ext_vector_type(4)))  unsigned int    u32x4;
typedef __attribute__((ext_vector_type(8)))  int             i32x8;
typedef __attribute__((ext_vector_type(4)))  int             i32x4;

union BFV { v16bf v; unsigned short u[16]; };

__device__ __forceinline__ unsigned short f2bf(float f) {
  unsigned int x = __float_as_uint(f);
  x += 0x7FFFu + ((x >> 16) & 1u);          // round-to-nearest-even
  return (unsigned short)(x >> 16);
}
__device__ __forceinline__ float bf2f(unsigned short h) {
  return __uint_as_float(((unsigned int)h) << 16);
}

// ---------- TDM: 2D tile (tileK x tileN, 2-byte elems) global -> LDS ----------
// D# packing per CDNA5 ISA ch.8 (group0 128b, group1 256b; groups 2/3 zero => 2D).
// This toolchain exposes the 6-arg builtin: (g0, g1, g2, g3, g4, cpol).
__device__ __forceinline__ void tdm_load_2d_bf16(
    unsigned ldsAddr, unsigned long long gaddr,
    unsigned tileK, unsigned tileN, unsigned long long strideElems)
{
  u32x4 g0;
  g0[0] = 1u;                                             // count=1, user desc
  g0[1] = ldsAddr;                                        // lds_addr (bytes)
  g0[2] = (unsigned)(gaddr & 0xFFFFFFFFu);                // global_addr[31:0]
  g0[3] = (unsigned)((gaddr >> 32) & 0x01FFFFFFu)         // global_addr[56:32]
        | (2u << 30);                                     // type = 2 (image)
  i32x8 g1;
  g1[0] = (int)(1u << 16);                                // data_size=1 (2B), mask=0
  g1[1] = (int)((tileK & 0xFFFFu) << 16);                 // tensor_dim0[15:0]
  g1[2] = (int)((tileK >> 16) | ((tileN & 0xFFFFu) << 16)); // dim0 hi | dim1 lo
  g1[3] = (int)((tileN >> 16) | ((tileK & 0xFFFFu) << 16)); // dim1 hi | tile_dim0
  g1[4] = (int)(tileN & 0xFFFFu);                         // tile_dim1 (tile_dim2=0)
  g1[5] = (int)(strideElems & 0xFFFFFFFFu);               // dim0_stride[31:0]
  g1[6] = (int)((strideElems >> 32) & 0xFFFFu);           // dim0_stride[47:32]
  g1[7] = 0;
  i32x4 z4 = { 0, 0, 0, 0 };
  i32x8 z8 = { 0, 0, 0, 0, 0, 0, 0, 0 };
  __builtin_amdgcn_tensor_load_to_lds(g0, g1, z4, z4, z8, 0);
}

// ---------- fp32 -> bf16 flat conversion ----------
__global__ __launch_bounds__(256) void cvt_f32_bf16_kernel(
    const float* __restrict__ src, unsigned short* __restrict__ dst, int n) {
  int i = (blockIdx.x * blockDim.x + threadIdx.x) * 4;
  if (i + 3 < n) {
    float4 f = *(const float4*)(src + i);
    us4 o = { f2bf(f.x), f2bf(f.y), f2bf(f.z), f2bf(f.w) };
    *(us4*)(dst + i) = o;
  }
}

// ---------- fp32 (1024x1024, [k][n]) -> bf16 transposed ([n][k]) ----------
__global__ __launch_bounds__(256) void cvt_transpose_kernel(
    const float* __restrict__ src, unsigned short* __restrict__ dst) {
  __shared__ unsigned short tile[32][33];
  const int tx = threadIdx.x & 31;
  const int ty = threadIdx.x >> 5;          // 0..7
  const int n0 = (blockIdx.x & 31) * 32;
  const int k0 = (blockIdx.x >> 5) * 32;
  #pragma unroll
  for (int r = 0; r < 32; r += 8)
    tile[ty + r][tx] = f2bf(src[(long long)(k0 + ty + r) * 1024 + n0 + tx]);
  __syncthreads();
  #pragma unroll
  for (int r = 0; r < 32; r += 8)
    dst[(long long)(n0 + ty + r) * 1024 + k0 + tx] = tile[tx][ty + r];
}

// ---------- direct-load WMMA GEMM (small/batched cases), B stored N x K ----------
// ABF16: A bf16 (else fp32 converted inline).
// CMODE: 0 = fp32 row-major, 1 = bf16 row-major, 2 = bf16 transposed store.
template <int ABF16, int CMODE, int HASA2>
__global__ __launch_bounds__(256) void wmma_gemm_nt_kernel(
    const void* __restrict__ Av, long long lda, long long strideA,
    const unsigned short* __restrict__ A2, long long lda2,
    const unsigned short* __restrict__ Bu, long long ldb, long long strideB,
    void* __restrict__ Cv, long long ldc, long long strideC,
    int tilesM, int tilesN, int nBatch, int K, float scale)
{
  const int wg = blockIdx.x * (blockDim.x >> 5) + ((int)threadIdx.x >> 5);
  const int total = nBatch * tilesM * tilesN;
  if (wg >= total) return;
  const int b   = wg / (tilesM * tilesN);
  const int rem = wg - b * (tilesM * tilesN);
  const int mt  = rem / tilesN;
  const int nt  = rem - mt * tilesN;
  const long long m0 = (long long)mt * 16;
  const long long n0 = (long long)nt * 64;
  const int lane   = threadIdx.x & 31;
  const int laneLo = lane & 15;
  const int laneHi = lane >> 4;

  const float*          Af = (const float*)Av;
  const unsigned short* Au = (const unsigned short*)Av;
  float*          Cf = (float*)Cv;
  unsigned short* Cu = (unsigned short*)Cv;

  v8f acc[4] = { {}, {}, {}, {} };

  const long long aRow   = m0 + laneLo;
  const long long aBase  = (long long)b * strideA + aRow * lda;
  const long long a2Base = HASA2 ? (aRow >> 3) * lda2 : 0;
  const unsigned short* bBase = Bu + (long long)b * strideB + (n0 + laneLo) * ldb;

  for (int kb = 0; kb < K; kb += 32) {
    BFV af;
    const int k0 = kb + laneHi * 8;
    const int k1 = kb + 16 + laneHi * 8;
    if (ABF16) {
      us8 c0 = *(const us8*)(Au + aBase + k0);
      us8 c1 = *(const us8*)(Au + aBase + k1);
      if (HASA2) {
        us8 d0 = *(const us8*)(A2 + a2Base + k0);
        us8 d1 = *(const us8*)(A2 + a2Base + k1);
        #pragma unroll
        for (int e = 0; e < 8; ++e) {
          af.u[e]     = f2bf(bf2f(c0[e]) + bf2f(d0[e]));
          af.u[e + 8] = f2bf(bf2f(c1[e]) + bf2f(d1[e]));
        }
      } else {
        #pragma unroll
        for (int e = 0; e < 8; ++e) { af.u[e] = c0[e]; af.u[e + 8] = c1[e]; }
      }
    } else {
      f8v c0 = *(const f8v*)(Af + aBase + k0);
      f8v c1 = *(const f8v*)(Af + aBase + k1);
      #pragma unroll
      for (int e = 0; e < 8; ++e) { af.u[e] = f2bf(c0[e]); af.u[e + 8] = f2bf(c1[e]); }
    }
    #pragma unroll
    for (int t = 0; t < 4; ++t) {
      const unsigned short* bp = bBase + (long long)t * 16 * ldb + kb + laneHi * 8;
      us8 c0 = *(const us8*)bp;
      us8 c1 = *(const us8*)(bp + 16);
      BFV bf;
      #pragma unroll
      for (int e = 0; e < 8; ++e) { bf.u[e] = c0[e]; bf.u[e + 8] = c1[e]; }
      acc[t] = __builtin_amdgcn_wmma_f32_16x16x32_bf16(
          false, af.v, false, bf.v, (short)0, acc[t], false, false);
    }
  }
  #pragma unroll
  for (int t = 0; t < 4; ++t) {
    const long long col = n0 + t * 16 + laneLo;
    if (CMODE == 2) {
      us8 w;
      #pragma unroll
      for (int r = 0; r < 8; ++r) w[r] = f2bf(acc[t][r] * scale);
      *(us8*)(Cu + (long long)b * strideC + col * ldc + m0 + laneHi * 8) = w;
    } else {
      #pragma unroll
      for (int r = 0; r < 8; ++r) {
        const long long row = m0 + laneHi * 8 + r;
        const long long off = (long long)b * strideC + row * ldc + col;
        const float val = acc[t][r] * scale;
        if (CMODE == 1) Cu[off] = f2bf(val);
        else            Cf[off] = val;
      }
    }
  }
}

// ---------- TDM-staged WMMA GEMM for the big 32768x1024x1024 GEMMs ----------
// Block tile: 128(M) x 64(N); 8 waves stacked in M share B tiles staged by the
// Tensor Data Mover into LDS (double-buffered 64x256 bf16 chunks).
// A is bf16 row-major; CBF16 selects output type; HASA2 fuses +A2[row>>3][k].
template <int CBF16, int HASA2>
__global__ __launch_bounds__(256) void wmma_gemm_nt_tdm_kernel(
    const unsigned short* __restrict__ Au, long long lda,
    const unsigned short* __restrict__ A2, long long lda2,
    const unsigned short* __restrict__ Bu, long long ldb,
    void* __restrict__ Cv, long long ldc,
    int tilesN, int K, float scale)
{
  __shared__ unsigned short ldsB[2][64 * 256];   // 2 x 32 KB
  const int wave   = (int)threadIdx.x >> 5;
  const int lane   = threadIdx.x & 31;
  const int laneLo = lane & 15;
  const int laneHi = lane >> 4;
  const int nt  = blockIdx.x % tilesN;
  const int mtB = blockIdx.x / tilesN;
  const long long m0 = (long long)mtB * 128 + wave * 16;
  const long long n0 = (long long)nt * 64;

  const long long aRow   = m0 + laneLo;
  const long long aBase  = aRow * lda;
  const long long a2Base = HASA2 ? (aRow >> 3) * lda2 : 0;

  const unsigned ldsBase = (unsigned)(unsigned long long)&ldsB[0][0];
  const unsigned short* bTile = Bu + n0 * ldb;

  v8f acc[4] = { {}, {}, {}, {} };

  const int nChunks = K >> 8;                    // K in chunks of 256
  if (wave == 0)
    tdm_load_2d_bf16(ldsBase, (unsigned long long)bTile, 256u, 64u,
                     (unsigned long long)ldb);

  for (int c = 0; c < nChunks; ++c) {
    const int buf = c & 1;
    if (wave == 0) {
      if (c + 1 < nChunks) {
        tdm_load_2d_bf16(ldsBase + (unsigned)(buf ^ 1) * (64u * 256u * 2u),
                         (unsigned long long)(bTile + (long long)(c + 1) * 256),
                         256u, 64u, (unsigned long long)ldb);
        __builtin_amdgcn_s_wait_tensorcnt(1);    // chunk c done (in-order)
      } else {
        __builtin_amdgcn_s_wait_tensorcnt(0);
      }
    }
    __syncthreads();                             // chunk c visible to all waves
    const unsigned short* Bl = &ldsB[buf][0];
    #pragma unroll 2
    for (int kk = 0; kk < 256; kk += 32) {
      const int kb = c * 256 + kk;
      BFV af;
      const int k0 = kb + laneHi * 8;
      const int k1 = kb + 16 + laneHi * 8;
      us8 c0 = *(const us8*)(Au + aBase + k0);
      us8 c1 = *(const us8*)(Au + aBase + k1);
      if (HASA2) {
        us8 d0 = *(const us8*)(A2 + a2Base + k0);
        us8 d1 = *(const us8*)(A2 + a2Base + k1);
        #pragma unroll
        for (int e = 0; e < 8; ++e) {
          af.u[e]     = f2bf(bf2f(c0[e]) + bf2f(d0[e]));
          af.u[e + 8] = f2bf(bf2f(c1[e]) + bf2f(d1[e]));
        }
      } else {
        #pragma unroll
        for (int e = 0; e < 8; ++e) { af.u[e] = c0[e]; af.u[e + 8] = c1[e]; }
      }
      #pragma unroll
      for (int t = 0; t < 4; ++t) {
        const unsigned short* bp = Bl + (t * 16 + laneLo) * 256 + kk + laneHi * 8;
        us8 b0 = *(const us8*)bp;
        us8 b1 = *(const us8*)(bp + 16);
        BFV bf;
        #pragma unroll
        for (int e = 0; e < 8; ++e) { bf.u[e] = b0[e]; bf.u[e + 8] = b1[e]; }
        acc[t] = __builtin_amdgcn_wmma_f32_16x16x32_bf16(
            false, af.v, false, bf.v, (short)0, acc[t], false, false);
      }
    }
    __syncthreads();                             // buffer reuse hand-off
  }

  float* Cf = (float*)Cv;
  unsigned short* Cu = (unsigned short*)Cv;
  #pragma unroll
  for (int t = 0; t < 4; ++t) {
    const long long col = n0 + t * 16 + laneLo;
    #pragma unroll
    for (int r = 0; r < 8; ++r) {
      const long long row = m0 + laneHi * 8 + r;
      const float val = acc[t][r] * scale;
      if (CBF16) Cu[row * ldc + col] = f2bf(val);
      else       Cf[row * ldc + col] = val;
    }
  }
}

// ---------- per-cube (8x8) attention + mean-pool (reps -> bf16) ----------
__global__ __launch_bounds__(256) void cube_attn_kernel(
    const unsigned short* __restrict__ q1, const unsigned short* __restrict__ k1,
    const unsigned short* __restrict__ v1, unsigned short* __restrict__ out1,
    unsigned short* __restrict__ reps, float scale)
{
  __shared__ float sPart[64][4];
  __shared__ float sP[8][8];
  const long long m0 = (long long)blockIdx.x * 8;
  const int t = threadIdx.x;
  {
    const int pid = t >> 2, part = t & 3;
    const int i = pid >> 3, j = pid & 7;
    const unsigned short* qr = q1 + (m0 + i) * 1024 + part * 256;
    const unsigned short* kr = k1 + (m0 + j) * 1024 + part * 256;
    float s = 0.f;
    for (int c = 0; c < 256; c += 8) {
      us8 qa = *(const us8*)(qr + c);
      us8 ka = *(const us8*)(kr + c);
      #pragma unroll
      for (int e = 0; e < 8; ++e) s += bf2f(qa[e]) * bf2f(ka[e]);
    }
    sPart[pid][part] = s;
  }
  __syncthreads();
  if (t < 64) {
    float s = (sPart[t][0] + sPart[t][1] + sPart[t][2] + sPart[t][3]) * scale;
    sP[t >> 3][t & 7] = s;
  }
  __syncthreads();
  if (t < 8) {
    float mx = sP[t][0];
    #pragma unroll
    for (int j = 1; j < 8; ++j) mx = fmaxf(mx, sP[t][j]);
    float ex[8]; float sum = 0.f;
    #pragma unroll
    for (int j = 0; j < 8; ++j) { ex[j] = __expf(sP[t][j] - mx); sum += ex[j]; }
    const float inv = 1.f / sum;
    #pragma unroll
    for (int j = 0; j < 8; ++j) sP[t][j] = ex[j] * inv;
  }
  __syncthreads();
  const int d0 = t * 4;
  float o[8][4];
  #pragma unroll
  for (int i = 0; i < 8; ++i) { o[i][0] = o[i][1] = o[i][2] = o[i][3] = 0.f; }
  #pragma unroll
  for (int j = 0; j < 8; ++j) {
    us4 vv = *(const us4*)(v1 + (m0 + j) * 1024 + d0);
    const float v0 = bf2f(vv[0]), va = bf2f(vv[1]), vb = bf2f(vv[2]), vc = bf2f(vv[3]);
    #pragma unroll
    for (int i = 0; i < 8; ++i) {
      const float p = sP[i][j];
      o[i][0] += p * v0; o[i][1] += p * va; o[i][2] += p * vb; o[i][3] += p * vc;
    }
  }
  #pragma unroll
  for (int i = 0; i < 8; ++i) {
    us4 w = { f2bf(o[i][0]), f2bf(o[i][1]), f2bf(o[i][2]), f2bf(o[i][3]) };
    *(us4*)(out1 + (m0 + i) * 1024 + d0) = w;
  }
  float r0 = 0, r1 = 0, r2 = 0, r3 = 0;
  #pragma unroll
  for (int i = 0; i < 8; ++i) { r0 += o[i][0]; r1 += o[i][1]; r2 += o[i][2]; r3 += o[i][3]; }
  us4 rw = { f2bf(r0 * 0.125f), f2bf(r1 * 0.125f), f2bf(r2 * 0.125f), f2bf(r3 * 0.125f) };
  *(us4*)(reps + (long long)blockIdx.x * 1024 + d0) = rw;
}

// ---------- 512-wide row softmax (fp32 in, fp32 out) ----------
__global__ __launch_bounds__(256) void softmax512_kernel(
    const float* __restrict__ S, float* __restrict__ P)
{
  __shared__ float red[256];
  const long long row = blockIdx.x;
  const float* s = S + row * 512;
  float*       p = P + row * 512;
  const int t = threadIdx.x;
  const float a = s[t], b = s[t + 256];
  red[t] = fmaxf(a, b);
  __syncthreads();
  for (int off = 128; off > 0; off >>= 1) {
    if (t < off) red[t] = fmaxf(red[t], red[t + off]);
    __syncthreads();
  }
  const float mx = red[0];
  __syncthreads();
  const float e0 = __expf(a - mx), e1 = __expf(b - mx);
  red[t] = e0 + e1;
  __syncthreads();
  for (int off = 128; off > 0; off >>= 1) {
    if (t < off) red[t] += red[t + off];
    __syncthreads();
  }
  const float inv = 1.f / red[0];
  p[t] = e0 * inv;
  p[t + 256] = e1 * inv;
}

// ---------- host ----------
extern "C" void kernel_launch(void* const* d_in, const int* in_sizes, int n_in,
                              void* d_out, int out_size, void* d_ws, size_t ws_size,
                              hipStream_t stream) {
  const float SCALE = 0.0883883476483184f;   // 128^-0.5
  const size_t MB = 1024ull * 1024ull;
  const float* x = (const float*)d_in[0];
  char* ws = (char*)d_ws;

  unsigned short* wT[7];                                    // transposed bf16 weights
  for (int i = 0; i < 7; ++i) wT[i] = (unsigned short*)(ws + (size_t)i * 2 * MB);
  unsigned short* xb   = (unsigned short*)(ws + 14 * MB);   // 64 MB bf16
  unsigned short* q1   = (unsigned short*)(ws + 78 * MB);   // 64 MB bf16
  unsigned short* k1   = (unsigned short*)(ws + 142 * MB);  // 64 MB bf16
  unsigned short* v1   = (unsigned short*)(ws + 206 * MB);  // 64 MB bf16
  unsigned short* o1   = (unsigned short*)(ws + 270 * MB);  // 64 MB bf16
  unsigned short* reps = (unsigned short*)(ws + 334 * MB);  // 8 MB bf16
  float*          q2   = (float*)(ws + 342 * MB);           // 16 MB f32
  unsigned short* k2   = (unsigned short*)(ws + 358 * MB);  // 8 MB bf16
  unsigned short* v2t  = (unsigned short*)(ws + 366 * MB);  // 8 MB bf16 (1024x4096 ^T)
  float*          s2   = (float*)(ws + 374 * MB);           // 8 MB f32
  float*          p2   = (float*)(ws + 382 * MB);           // 8 MB f32
  unsigned short* itr  = (unsigned short*)(ws + 390 * MB);  // 8 MB bf16

  const unsigned short* nil = (const unsigned short*)nullptr;

  // 1) weights -> bf16 transposed; x -> bf16
  for (int i = 0; i < 7; ++i)
    cvt_transpose_kernel<<<1024, 256, 0, stream>>>((const float*)d_in[1 + i], wT[i]);
  cvt_f32_bf16_kernel<<<32768, 256, 0, stream>>>(x, xb, 32768 * 1024);

  // 2) Level-1 QKV (TDM-staged): (32768x1024) @ W, block 128x64
  wmma_gemm_nt_tdm_kernel<1,0><<<4096, 256, 0, stream>>>(
      xb, 1024, nil, 0, wT[0], 1024, q1, 1024, 16, 1024, 1.f);
  wmma_gemm_nt_tdm_kernel<1,0><<<4096, 256, 0, stream>>>(
      xb, 1024, nil, 0, wT[1], 1024, k1, 1024, 16, 1024, 1.f);
  wmma_gemm_nt_tdm_kernel<1,0><<<4096, 256, 0, stream>>>(
      xb, 1024, nil, 0, wT[2], 1024, v1, 1024, 16, 1024, 1.f);

  // 3) intra-cube attention + mean pooling (reps bf16)
  cube_attn_kernel<<<4096, 256, 0, stream>>>(q1, k1, v1, o1, reps, SCALE);

  // 4) Level-2 QKV: (4096x1024) @ W (direct-load GEMM)
  wmma_gemm_nt_kernel<1,0,0><<<512, 256, 0, stream>>>(
      reps, 1024, 0, nil, 0, wT[3], 1024, 0, q2, 1024, 0, 256, 16, 1, 1024, 1.f);
  wmma_gemm_nt_kernel<1,1,0><<<512, 256, 0, stream>>>(
      reps, 1024, 0, nil, 0, wT[4], 1024, 0, k2, 1024, 0, 256, 16, 1, 1024, 1.f);
  wmma_gemm_nt_kernel<1,2,0><<<512, 256, 0, stream>>>(    // v2 stored transposed
      reps, 1024, 0, nil, 0, wT[5], 1024, 0, v2t, 4096, 0, 256, 16, 1, 1024, 1.f);

  // 5) scores2 = SCALE * q2 @ k2^T (k2 naturally NxK), batched over 8
  wmma_gemm_nt_kernel<0,0,0><<<256, 256, 0, stream>>>(
      q2, 1024, 512 * 1024, nil, 0, k2, 1024, 512 * 1024,
      s2, 512, 512 * 512, 32, 8, 8, 1024, SCALE);

  // 6) softmax over 512-wide rows (8*512 rows)
  softmax512_kernel<<<4096, 256, 0, stream>>>(s2, p2);

  // 7) inter = P @ v2  (B = v2t, NxK with ldb=4096, batch offset 512)
  wmma_gemm_nt_kernel<0,1,0><<<512, 256, 0, stream>>>(
      p2, 512, 512 * 512, nil, 0, v2t, 4096, 512,
      itr, 1024, 512 * 1024, 32, 16, 8, 512, 1.f);

  // 8) out = (out1 + inter[m/8]) @ Wo (TDM-staged), C fp32 -> d_out
  wmma_gemm_nt_tdm_kernel<0,1><<<4096, 256, 0, stream>>>(
      o1, 1024, itr, 1024, wT[6], 1024,
      (float*)d_out, 1024, 16, 1024, 1.f);
}
